// PhaseCoupledComplexAttention_24902220382734
// MI455X (gfx1250) — matrix operand order repaired and checked by reading, not compiled, not run on hardware
//
#include <hip/hip_runtime.h>
#include <cstdint>

// Problem constants (must match reference)
#define B_   2
#define S_   2048
#define D_   1024
#define H_   16
#define HD_  64
#define NELEM (B_*S_*D_)    // 4194304
#define WELEM (D_*D_)       // 1048576

typedef unsigned short u16;
typedef __attribute__((ext_vector_type(16))) __bf16 v16bf;
typedef __attribute__((ext_vector_type(8)))  float  v8f;

struct __attribute__((aligned(16))) U4 { unsigned int x, y, z, w; };

union Frag {
    U4 q[2];
    unsigned int u[8];
    v16bf v;
};

__device__ inline u16 f2bf(float f) {
    unsigned int u = __float_as_uint(f);
    u += 0x7FFFu + ((u >> 16) & 1u);       // round-to-nearest-even
    return (u16)(u >> 16);
}

__device__ inline v8f vzero() {
    v8f v;
#pragma unroll
    for (int i = 0; i < 8; ++i) v[i] = 0.f;
    return v;
}

__device__ inline v8f bf16_wmma(const Frag& a, const Frag& b, v8f c) {
    // D = A(16x32 bf16) * B(32x16 bf16) + C(16x16 f32)
    return __builtin_amdgcn_wmma_f32_16x16x32_bf16(false, a.v, false, b.v,
                                                   (short)0, c, false, false);
}

// Low 32 bits of a generic pointer to LDS == byte offset into workgroup LDS.
__device__ inline unsigned lds_addr(const void* p) {
    return (unsigned)(unsigned long long)p;
}

// CDNA5 async DMA: global -> LDS, 16B per lane, tracked by ASYNCcnt.
// GVS mode: 64-bit SGPR base + 32-bit per-lane byte offset.
__device__ inline void async_b128(unsigned ldsOff, unsigned gOff, const u16* base) {
    asm volatile("global_load_async_to_lds_b128 %0, %1, %2"
                 :: "v"(ldsOff), "v"(gOff), "s"(base)
                 : "memory");
}

__device__ inline void wait_async0() {
    asm volatile("s_wait_asynccnt 0x0" ::: "memory");
}

// ---------------------------------------------------------------------------
// f32 -> bf16 conversion
// ---------------------------------------------------------------------------
__global__ __launch_bounds__(256) void cvt_bf16(const float* __restrict__ in,
                                                u16* __restrict__ out, int n) {
    int i = blockIdx.x * 256 + threadIdx.x;
    if (i < n) out[i] = f2bf(in[i]);
}

// ---------------------------------------------------------------------------
// v[b,s,h*64+d] -> vt[((b*H+h)*64+d)*S + s]
// ---------------------------------------------------------------------------
__global__ __launch_bounds__(256) void transpose_v(const u16* __restrict__ v,
                                                   u16* __restrict__ vt) {
    int idx = blockIdx.x * 256 + threadIdx.x;   // ((b*H+h)*HD + d)*S + s
    int s = idx & (S_ - 1);
    int t = idx >> 11;
    int d = t & (HD_ - 1);
    t >>= 6;
    int h = t & (H_ - 1);
    int b = t >> 4;
    vt[idx] = v[((size_t)(b * S_ + s)) * D_ + h * HD_ + d];
}

// ---------------------------------------------------------------------------
// Complex GEMM: yr = Ar*Wr^T - Ai*Wi^T ; yi = Ai*Wr^T + Ar*Wi^T
// A: [M,K] bf16 row-major, W: [N,K] bf16 row-major.
// Double-buffered async global->LDS staging; xi*Wi goes to a separate
// accumulator (accS) and is subtracted in the epilogue (no per-iter xors).
// ---------------------------------------------------------------------------
#define BM 128
#define BN 64
#define BK 32
#define LSTR 40   // padded LDS row stride in u16 elements

__global__ __launch_bounds__(256) void complex_gemm(
    const u16* __restrict__ Ar, const u16* __restrict__ Ai,
    const u16* __restrict__ Wr, const u16* __restrict__ Wi,
    u16* __restrict__ oRb, u16* __restrict__ oIb,
    float* __restrict__ oRf, float* __restrict__ oIf,
    int M, int N, int K)
{
    __shared__ u16 sAr[2][BM * LSTR], sAi[2][BM * LSTR];
    __shared__ u16 sWr[2][BN * LSTR], sWi[2][BN * LSTR];

    const int tid   = threadIdx.x;
    const int lane  = tid & 31;
    const int wid   = tid >> 5;
    const int waveM = wid & 3;       // 4 waves along M (32 rows each)
    const int waveN = wid >> 2;      // 2 waves along N (32 cols each)
    const int bn = blockIdx.x, bm = blockIdx.y;

    v8f accR[2][2], accS[2][2], accI[2][2];
#pragma unroll
    for (int i = 0; i < 2; ++i)
#pragma unroll
        for (int j = 0; j < 2; ++j) {
            accR[i][j] = vzero(); accS[i][j] = vzero(); accI[i][j] = vzero();
        }

    const int arow = tid >> 1;            // 0..127
    const int acol = (tid & 1) * 16;      // 0 / 16 elements
    const int wrow = tid >> 2;            // 0..63
    const int wcol = (tid & 3) * 8;       // 0/8/16/24 elements

    const unsigned aoffB = (unsigned)(((bm * BM + arow) * K + acol) * 2);
    const unsigned woffB = (unsigned)(((bn * BN + wrow) * K + wcol) * 2);

    unsigned lAr[2], lAi[2], lWr[2], lWi[2];
#pragma unroll
    for (int st = 0; st < 2; ++st) {
        lAr[st] = lds_addr(&sAr[st][arow * LSTR + acol]);
        lAi[st] = lds_addr(&sAi[st][arow * LSTR + acol]);
        lWr[st] = lds_addr(&sWr[st][wrow * LSTR + wcol]);
        lWi[st] = lds_addr(&sWi[st][wrow * LSTR + wcol]);
    }

    auto issue = [&](int st, int kb) {
        const unsigned kb2 = (unsigned)(kb * 2);
        async_b128(lAr[st],      aoffB + kb2,      Ar);
        async_b128(lAr[st] + 16, aoffB + kb2 + 16, Ar);
        async_b128(lAi[st],      aoffB + kb2,      Ai);
        async_b128(lAi[st] + 16, aoffB + kb2 + 16, Ai);
        async_b128(lWr[st],      woffB + kb2,      Wr);
        async_b128(lWi[st],      woffB + kb2,      Wi);
    };

    issue(0, 0);

    const int hA = (lane & 16) ? 8 : 0;     // A frag: K half select
    const int kB = (lane & 16) ? 8 : 0;     // B frag: K half select

    for (int kb = 0; kb < K; kb += BK) {
        const int st = (kb >> 5) & 1;
        wait_async0();        // my async writes for this stage are in LDS
        __syncthreads();      // everyone's writes done; prev compute done
        if (kb + BK < K) issue(st ^ 1, kb + BK);

        Frag ar[2], ai[2], wr[2], wi[2];
#pragma unroll
        for (int mt = 0; mt < 2; ++mt) {
            int row  = waveM * 32 + mt * 16 + (lane & 15);
            int base = row * LSTR + hA;
            ar[mt].q[0] = *(const U4*)&sAr[st][base];
            ar[mt].q[1] = *(const U4*)&sAr[st][base + 16];
            ai[mt].q[0] = *(const U4*)&sAi[st][base];
            ai[mt].q[1] = *(const U4*)&sAi[st][base + 16];
        }
#pragma unroll
        for (int nt = 0; nt < 2; ++nt) {
            int row  = waveN * 32 + nt * 16 + (lane & 15);
            int base = row * LSTR + kB;
            wr[nt].q[0] = *(const U4*)&sWr[st][base];
            wr[nt].q[1] = *(const U4*)&sWr[st][base + 16];
            wi[nt].q[0] = *(const U4*)&sWi[st][base];
            wi[nt].q[1] = *(const U4*)&sWi[st][base + 16];
        }
#pragma unroll
        for (int mt = 0; mt < 2; ++mt)
#pragma unroll
            for (int nt = 0; nt < 2; ++nt) {
                accR[mt][nt] = bf16_wmma(ar[mt], wr[nt], accR[mt][nt]);
                accS[mt][nt] = bf16_wmma(ai[mt], wi[nt], accS[mt][nt]);
                accI[mt][nt] = bf16_wmma(ai[mt], wr[nt], accI[mt][nt]);
                accI[mt][nt] = bf16_wmma(ar[mt], wi[nt], accI[mt][nt]);
            }
    }

    const int rowOff = (lane & 16) ? 8 : 0;
#pragma unroll
    for (int mt = 0; mt < 2; ++mt)
#pragma unroll
        for (int nt = 0; nt < 2; ++nt)
#pragma unroll
            for (int r = 0; r < 8; ++r) {
                size_t row = (size_t)bm * BM + waveM * 32 + mt * 16 + r + rowOff;
                size_t col = (size_t)bn * BN + waveN * 32 + nt * 16 + (lane & 15);
                float vr = accR[mt][nt][r] - accS[mt][nt][r];   // Re: xr*wr - xi*wi
                float vi = accI[mt][nt][r];                     // Im: xi*wr + xr*wi
                vr = fminf(fmaxf(vr, -1e4f), 1e4f);
                vi = fminf(fmaxf(vi, -1e4f), 1e4f);
                size_t idx = row * (size_t)N + col;
                if (oRf) { oRf[idx] = vr; oIf[idx] = vi; }
                else     { oRb[idx] = f2bf(vr); oIb[idx] = f2bf(vi); }
            }
}

// ---------------------------------------------------------------------------
// Phase-coupled flash attention. Grid = B*H*(S/64), block = 128 (4 waves).
// Each wave owns a 16-row query tile; loops over 32-key blocks with online
// softmax on |Q conj(K)| scores.
// ---------------------------------------------------------------------------
__global__ __launch_bounds__(128) void phase_attn(
    const u16* __restrict__ qr, const u16* __restrict__ qi,
    const u16* __restrict__ kr, const u16* __restrict__ ki,
    const u16* __restrict__ vtr, const u16* __restrict__ vti,
    u16* __restrict__ outr, u16* __restrict__ outi)
{
    __shared__ u16 pLds[4 * 16 * LSTR];    // per-wave 16x32 prob staging (padded)

    const int lane = threadIdx.x & 31;
    const int wid  = threadIdx.x >> 5;
    const int blk  = blockIdx.x;
    const int qt = blk & 31;
    const int h  = (blk >> 5) & 15;
    const int b  = blk >> 9;
    const int qBase = qt * 64 + wid * 16;

    const size_t zoff = (size_t)b * S_ * D_ + h * HD_;
    const u16* qrh = qr + zoff;  const u16* qih = qi + zoff;
    const u16* krh = kr + zoff;  const u16* kih = ki + zoff;
    const size_t voff = (size_t)(b * H_ + h) * HD_ * S_;
    const u16* vtrh = vtr + voff; const u16* vtih = vti + voff;

    const int m15 = lane & 15;
    const int hA = (lane & 16) ? 8 : 0;
    const int kB = (lane & 16) ? 8 : 0;

    // Q fragments (16x64 -> 2 k-steps)
    Frag qrA[2], qiA[2];
#pragma unroll
    for (int ks = 0; ks < 2; ++ks) {
        const u16* p0 = qrh + (size_t)(qBase + m15) * D_ + ks * 32 + hA;
        qrA[ks].q[0] = *(const U4*)p0;
        qrA[ks].q[1] = *(const U4*)(p0 + 16);
        const u16* p1 = qih + (size_t)(qBase + m15) * D_ + ks * 32 + hA;
        qiA[ks].q[0] = *(const U4*)p1;
        qiA[ks].q[1] = *(const U4*)(p1 + 16);
    }

    float mrow[8], lrow[8];
    v8f oR[4], oI[4];
#pragma unroll
    for (int r = 0; r < 8; ++r) { mrow[r] = -1e30f; lrow[r] = 0.f; }
#pragma unroll
    for (int d = 0; d < 4; ++d) { oR[d] = vzero(); oI[d] = vzero(); }

    u16* pw = &pLds[wid * 16 * LSTR];
    const int rowBase = (lane & 16) ? 8 : 0;

    for (int kb = 0; kb < S_; kb += 32) {
        if (kb + 32 < S_) {   // prefetch next K/V tiles (global_prefetch_b8)
            __builtin_prefetch(krh + (size_t)(kb + 32 + m15) * D_, 0, 3);
            __builtin_prefetch(kih + (size_t)(kb + 32 + m15) * D_, 0, 3);
            __builtin_prefetch(vtrh + (size_t)m15 * S_ + kb + 32, 0, 3);
            __builtin_prefetch(vtih + (size_t)m15 * S_ + kb + 32, 0, 3);
        }
        // scores for two 16-key halves; si accumulated as (QiKr) - (QrKi)
        v8f sR[2], sIa[2], sIb[2];
#pragma unroll
        for (int hc = 0; hc < 2; ++hc) { sR[hc] = vzero(); sIa[hc] = vzero(); sIb[hc] = vzero(); }
#pragma unroll
        for (int hc = 0; hc < 2; ++hc) {
#pragma unroll
            for (int ks = 0; ks < 2; ++ks) {
                Frag kR, kI;
                const u16* pk = krh + (size_t)(kb + hc * 16 + m15) * D_ + ks * 32 + kB;
                kR.q[0] = *(const U4*)pk;  kR.q[1] = *(const U4*)(pk + 16);
                const u16* pki = kih + (size_t)(kb + hc * 16 + m15) * D_ + ks * 32 + kB;
                kI.q[0] = *(const U4*)pki; kI.q[1] = *(const U4*)(pki + 16);
                sR[hc]  = bf16_wmma(qrA[ks], kR, sR[hc]);
                sR[hc]  = bf16_wmma(qiA[ks], kI, sR[hc]);
                sIa[hc] = bf16_wmma(qiA[ks], kR, sIa[hc]);
                sIb[hc] = bf16_wmma(qrA[ks], kI, sIb[hc]);
            }
        }
        // magnitude scores: clamp +/-250, hypot, * HD^-0.5
        float sc0[8], sc1[8];
#pragma unroll
        for (int r = 0; r < 8; ++r) {
            float a0 = fminf(fmaxf(sR[0][r], -250.f), 250.f);
            float c0 = fminf(fmaxf(sIa[0][r] - sIb[0][r], -250.f), 250.f);
            sc0[r] = sqrtf(a0 * a0 + c0 * c0) * 0.125f;
            float a1 = fminf(fmaxf(sR[1][r], -250.f), 250.f);
            float c1 = fminf(fmaxf(sIa[1][r] - sIb[1][r], -250.f), 250.f);
            sc1[r] = sqrtf(a1 * a1 + c1 * c1) * 0.125f;
        }
        // online softmax (row reductions across 16-lane halves)
#pragma unroll
        for (int r = 0; r < 8; ++r) {
            float t = fmaxf(sc0[r], sc1[r]);
#pragma unroll
            for (int off = 1; off < 16; off <<= 1) t = fmaxf(t, __shfl_xor(t, off, 32));
            float mn = fmaxf(mrow[r], t);
            float alpha = __expf(mrow[r] - mn);
            float p0 = __expf(sc0[r] - mn);
            float p1 = __expf(sc1[r] - mn);
            float rs = p0 + p1;
#pragma unroll
            for (int off = 1; off < 16; off <<= 1) rs += __shfl_xor(rs, off, 32);
            lrow[r] = lrow[r] * alpha + rs;
            mrow[r] = mn;
#pragma unroll
            for (int d = 0; d < 4; ++d) { oR[d][r] *= alpha; oI[d][r] *= alpha; }
            pw[(rowBase + r) * LSTR + m15]      = f2bf(p0);
            pw[(rowBase + r) * LSTR + 16 + m15] = f2bf(p1);
        }
        // Per-wave private LDS: same-wave DS ops are in-order; only a
        // compiler ordering barrier is needed (no workgroup barrier).
        asm volatile("" ::: "memory");
        Frag pA;   // probs as A fragment (C-layout -> A-layout via LDS)
        {
            const u16* pp = &pw[m15 * LSTR + hA];
            pA.q[0] = *(const U4*)pp;
            pA.q[1] = *(const U4*)(pp + 16);
        }
        asm volatile("" ::: "memory");
#pragma unroll
        for (int d = 0; d < 4; ++d) {
            Frag vR, vI;
            const u16* pv = vtrh + (size_t)(d * 16 + m15) * S_ + kb + kB;
            vR.q[0] = *(const U4*)pv;  vR.q[1] = *(const U4*)(pv + 16);
            const u16* pvi = vtih + (size_t)(d * 16 + m15) * S_ + kb + kB;
            vI.q[0] = *(const U4*)pvi; vI.q[1] = *(const U4*)(pvi + 16);
            oR[d] = bf16_wmma(pA, vR, oR[d]);
            oI[d] = bf16_wmma(pA, vI, oI[d]);
        }
    }

    // normalize, clamp (_safe), store bf16 [B,S,D]
#pragma unroll
    for (int r = 0; r < 8; ++r) {
        float inv = 1.f / lrow[r];
#pragma unroll
        for (int d = 0; d < 4; ++d) {
            float vr = fminf(fmaxf(oR[d][r] * inv, -1e4f), 1e4f);
            float vi = fminf(fmaxf(oI[d][r] * inv, -1e4f), 1e4f);
            size_t row = (size_t)b * S_ + qBase + rowBase + r;
            size_t col = (size_t)h * HD_ + d * 16 + m15;
            outr[row * D_ + col] = f2bf(vr);
            outi[row * D_ + col] = f2bf(vi);
        }
    }
}

// ---------------------------------------------------------------------------
extern "C" void kernel_launch(void* const* d_in, const int* in_sizes, int n_in,
                              void* d_out, int out_size, void* d_ws, size_t ws_size,
                              hipStream_t stream) {
    (void)in_sizes; (void)n_in; (void)out_size; (void)ws_size;

    const float* zr = (const float*)d_in[0];
    const float* zi = (const float*)d_in[1];
    const float* w_in[8];
    for (int i = 0; i < 8; ++i) w_in[i] = (const float*)d_in[2 + i];

    char* ws = (char*)d_ws;
    const size_t ZB = (size_t)NELEM * 2;   // 8 MB per bf16 activation buffer
    const size_t WB = (size_t)WELEM * 2;   // 2 MB per bf16 weight matrix

    u16* zr_bf = (u16*)(ws);
    u16* zi_bf = (u16*)(ws + ZB);
    u16* wbf[8];
    for (int i = 0; i < 8; ++i) wbf[i] = (u16*)(ws + 2 * ZB + (size_t)i * WB);
    char* p = ws + 2 * ZB + 8 * WB;
    u16* q_r  = (u16*)(p);           u16* q_i  = (u16*)(p + ZB);
    u16* k_r  = (u16*)(p + 2 * ZB);  u16* k_i  = (u16*)(p + 3 * ZB);
    u16* v_r  = (u16*)(p + 4 * ZB);  u16* v_i  = (u16*)(p + 5 * ZB);
    u16* vt_r = (u16*)(p + 6 * ZB);  u16* vt_i = (u16*)(p + 7 * ZB);
    u16* ao_r = (u16*)(p + 8 * ZB);  u16* ao_i = (u16*)(p + 9 * ZB);

    cvt_bf16<<<NELEM / 256, 256, 0, stream>>>(zr, zr_bf, NELEM);
    cvt_bf16<<<NELEM / 256, 256, 0, stream>>>(zi, zi_bf, NELEM);
    for (int i = 0; i < 8; ++i)
        cvt_bf16<<<WELEM / 256, 256, 0, stream>>>(w_in[i], wbf[i], WELEM);

    dim3 gg(D_ / BN, (B_ * S_) / BM);   // (16, 32)
    complex_gemm<<<gg, 256, 0, stream>>>(zr_bf, zi_bf, wbf[0], wbf[1],
                                         q_r, q_i, nullptr, nullptr,
                                         B_ * S_, D_, D_);
    complex_gemm<<<gg, 256, 0, stream>>>(zr_bf, zi_bf, wbf[2], wbf[3],
                                         k_r, k_i, nullptr, nullptr,
                                         B_ * S_, D_, D_);
    complex_gemm<<<gg, 256, 0, stream>>>(zr_bf, zi_bf, wbf[4], wbf[5],
                                         v_r, v_i, nullptr, nullptr,
                                         B_ * S_, D_, D_);

    transpose_v<<<NELEM / 256, 256, 0, stream>>>(v_r, vt_r);
    transpose_v<<<NELEM / 256, 256, 0, stream>>>(v_i, vt_i);

    phase_attn<<<B_ * H_ * (S_ / 64), 128, 0, stream>>>(q_r, q_i, k_r, k_i,
                                                        vt_r, vt_i, ao_r, ao_i);

    float* yr = (float*)d_out;
    float* yi = yr + NELEM;
    complex_gemm<<<gg, 256, 0, stream>>>(ao_r, ao_i, wbf[6], wbf[7],
                                         nullptr, nullptr, yr, yi,
                                         B_ * S_, D_, D_);
}
